// bottlefit_quantizer_8048768713433
// MI455X (gfx1250) — matrix operand order repaired and checked by reading, not compiled, not run on hardware
//
#include <hip/hip_runtime.h>
#include <hip/hip_bf16.h>

// ---------------------------------------------------------------------------
// BottleFit quantizer pipeline on gfx1250 (MI455X), fused as 3 WMMA GEMMs.
//   repack : W_* (f32, row-major)  -> f16 fragment-major chunks (one-time)
//   GEMM1  : Y = rint(sigmoid(x @ W_enc + b_enc) * 15) / 15   (f16 -> ws)
//   GEMM2  : H = relu(Y @ W_dec + b_dec)                      (f16 -> ws)
//   GEMM3  : P = H @ W_tail + b_tail                          (f32 -> d_out)
// The 4-bit pack/unpack round trip is the identity rint(y*15)/15.
// B tiles are 4KB contiguous fragment-major chunks, loaded into LDS by the
// Tensor Data Mover (tensor_load_to_lds + s_wait_tensorcnt); A tiles are
// staged through LDS in WMMA fragment order via ds_store_b128.
// ---------------------------------------------------------------------------

typedef _Float16 h8   __attribute__((ext_vector_type(8)));   // 16B: one *_b128
typedef _Float16 h16  __attribute__((ext_vector_type(16)));  // WMMA A/B fragment
typedef float    f8   __attribute__((ext_vector_type(8)));   // WMMA C/D fragment
typedef unsigned int u32x4 __attribute__((ext_vector_type(4)));
typedef int          i32x8 __attribute__((ext_vector_type(8)));
typedef int          i32x4 __attribute__((ext_vector_type(4)));

#define BM 128
#define BN 64
#define BK 32

#if defined(__has_builtin)
#if __has_builtin(__builtin_amdgcn_tensor_load_to_lds)
#define HAS_TDM 1
#endif
#endif
#ifndef HAS_TDM
#define HAS_TDM 0
#endif

#if HAS_TDM
// 1-D contiguous TDM copy: global -> LDS, `n8` units of 8 bytes.
// D# fields per CDNA5 ISA 8.3/8.4 (count=1, type=2, data_size=8B, 1-D tile).
// This toolchain exposes the 6-arg builtin:
//   (u32x4 group0, i32x8 group1, i32x4 group2, i32x4 group3, i32x8, i32 cpol)
__device__ __forceinline__ void tdm_load_1d(unsigned lds_byte_off,
                                            unsigned long long gaddr,
                                            unsigned n8)
{
    u32x4 g0;
    g0[0] = 1u;                                            // count=1, user mode
    g0[1] = lds_byte_off;                                  // lds_addr[31:0]
    g0[2] = (unsigned)(gaddr & 0xFFFFFFFFu);               // global_addr[31:0]
    g0[3] = (unsigned)((gaddr >> 32) & 0x01FFFFFFu)        // global_addr[56:32]
          | (2u << 30);                                    // type=2 ("image")
    i32x8 g1;
    g1[0] = (3 << 16);                                     // data_size=3 (8B)
    g1[1] = (int)(n8 << 16);                               // tensor_dim0[15:0]
    g1[2] = (int)((n8 >> 16) & 0xFFFFu) | (1 << 16);       // dim0[31:16], dim1=1
    g1[3] = (int)(n8 << 16);                               // tile_dim0
    g1[4] = 1;                                             // tile_dim1 = 1
    g1[5] = (int)n8;                                       // dim0_stride[31:0]
    g1[6] = 0;
    g1[7] = 0;
    const i32x4 z4 = {0, 0, 0, 0};
    const i32x8 z8 = {0, 0, 0, 0, 0, 0, 0, 0};
    __builtin_amdgcn_tensor_load_to_lds(g0, g1, z4, z4, z8, 0);
}
#endif

// ---------------------------------------------------------------------------
// One-time weight repack: W f32 [K,N] row-major -> Wf f16 fragment-major:
//   chunk(kt, nb) = 4 n-subtiles x 32 lanes x 16 elems  (4KB, contiguous)
//   lane = (n&15) + 16*((k>>4)&1) ; e = k&15           (contiguous k order, same as A staging)
// Pads columns N..Npad-1 with zeros (absorbs the N_cls=1000 tail).
// ---------------------------------------------------------------------------
__global__ __launch_bounds__(256)
void repack_w_kernel(const float* __restrict__ W, _Float16* __restrict__ Wf,
                     int K, int N, int Npad)
{
    const int t = blockIdx.x * 256 + threadIdx.x;
    if (t >= K * Npad) return;
    const int k = t / Npad;
    const int n = t - k * Npad;
    const float v = (n < N) ? W[(size_t)k * N + n] : 0.0f;

    const int kt = k >> 5, kk = k & 31;
    const int nb = n >> 6, ns = (n >> 4) & 3;
    const int lane = (n & 15) + ((kk >> 4) << 4);
    const int e    = kk & 15;
    const int NB   = Npad >> 6;
    Wf[((((size_t)kt * NB + nb) * 4 + ns) * 32 + lane) * 16 + e] = (_Float16)v;
}

// ---------------------------------------------------------------------------
// WMMA GEMM.  EPI: 0 = sigmoid + 4-bit quantize (f16 out)
//                  1 = relu (f16 out)
//                  2 = bias only, column-guarded stores (f32 out)
// A: row-major [M,K] (f32 if !A_F16, else f16).  Bf: fragment-major f16.
// ---------------------------------------------------------------------------
template <int EPI, bool A_F16>
__global__ __launch_bounds__(256)
void gemm_wmma_kernel(const void* __restrict__ Aptr,
                      const _Float16* __restrict__ Bf,
                      const float* __restrict__ bias, void* __restrict__ Cptr,
                      int N, int Npad, int K)
{
    __shared__ _Float16 ldsA[8 * 32 * 16];  // 8 m-subtiles, frag order : 8 KB
    __shared__ _Float16 ldsB[4 * 32 * 16];  // 4 n-subtiles, frag order : 4 KB

    const int tid  = threadIdx.x;
    const int lane = tid & 31;
    const int wave = tid >> 5;      // 8 waves
    const int wm   = wave >> 1;     // 0..3 : wave row (32 rows)
    const int wn   = wave & 1;      // 0..1 : wave col (32 cols)
    const int bm0  = blockIdx.y * BM;
    const int nb   = blockIdx.x;            // n-block (64 cols)
    const int bn0  = nb * BN;
    const int NB   = Npad >> 6;

    // A staging: thread -> (tile row, 16-wide k chunk), stored in frag order:
    //   lane = m + 16*(k>>4), e = k&15
    const int ar  = tid >> 1;            // 0..127
    const int akh = (tid & 1) * 16;      // 0 or 16
    const int as  = ar >> 4;
    const int am  = ar & 15;
    const int aeb = (akh >> 4) << 3;     // 0 or 8

    const unsigned ldsB_off = (unsigned)(size_t)(&ldsB[0]);  // LDS byte addr

    f8 acc[2][2] = {};

    const int KT = K / BK;
    for (int kt = 0; kt < KT; ++kt) {
        // ---------------- A: global -> regs (+ f32->f16 cvt) ----------------
        h8 a0, a1;
        if constexpr (A_F16) {
            const _Float16* Ag =
                (const _Float16*)Aptr + (size_t)(bm0 + ar) * K + kt * BK + akh;
            a0 = *(const h8*)(Ag + 0);
            a1 = *(const h8*)(Ag + 8);
        } else {
            const float* Ag =
                (const float*)Aptr + (size_t)(bm0 + ar) * K + kt * BK + akh;
            float4 q0 = *(const float4*)(Ag + 0);
            float4 q1 = *(const float4*)(Ag + 4);
            float4 q2 = *(const float4*)(Ag + 8);
            float4 q3 = *(const float4*)(Ag + 12);
            a0[0] = (_Float16)q0.x; a0[1] = (_Float16)q0.y;
            a0[2] = (_Float16)q0.z; a0[3] = (_Float16)q0.w;
            a0[4] = (_Float16)q1.x; a0[5] = (_Float16)q1.y;
            a0[6] = (_Float16)q1.z; a0[7] = (_Float16)q1.w;
            a1[0] = (_Float16)q2.x; a1[1] = (_Float16)q2.y;
            a1[2] = (_Float16)q2.z; a1[3] = (_Float16)q2.w;
            a1[4] = (_Float16)q3.x; a1[5] = (_Float16)q3.y;
            a1[6] = (_Float16)q3.z; a1[7] = (_Float16)q3.w;
        }
        if (kt + 1 < KT) {  // global_prefetch_b8 on next A tile
            if constexpr (A_F16) {
                __builtin_prefetch((const _Float16*)Aptr +
                                   (size_t)(bm0 + ar) * K + (kt + 1) * BK + akh, 0, 1);
            } else {
                __builtin_prefetch((const float*)Aptr +
                                   (size_t)(bm0 + ar) * K + (kt + 1) * BK + akh, 0, 1);
            }
        }

        __syncthreads();  // previous tile fully consumed (ldsA + ldsB free)

        // ---------------- B: TDM async copy of 4KB fragment chunk ----------
        const size_t chunk = ((size_t)kt * NB + nb) * (4 * 32 * 16);
#if HAS_TDM
        if (wave == 0) {
            tdm_load_1d(ldsB_off,
                        (unsigned long long)(size_t)(Bf + chunk),
                        (4 * 32 * 16 * 2) / 8 /* = 512 x 8B */);
        }
#else
        ((h8*)ldsB)[tid] = ((const h8*)(Bf + chunk))[tid];
#endif

        // ---------------- A: regs -> LDS in fragment order ------------------
        *(h8*)&ldsA[((as * 32 + am + akh) << 4) + 0] = a0;
        *(h8*)&ldsA[((as * 32 + am + akh) << 4) + 8] = a1;

#if HAS_TDM
        if (wave == 0) __builtin_amdgcn_s_wait_tensorcnt(0);
#endif
        __syncthreads();  // tile ready

        // ---------------- fragments + WMMA ----------------------------------
        h16 afrag[2], bfrag[2];
        #pragma unroll
        for (int i = 0; i < 2; ++i)
            afrag[i] = *(const h16*)&ldsA[((wm * 2 + i) * 32 + lane) << 4];
        #pragma unroll
        for (int j = 0; j < 2; ++j)
            bfrag[j] = *(const h16*)&ldsB[((wn * 2 + j) * 32 + lane) << 4];

        #pragma unroll
        for (int i = 0; i < 2; ++i)
            #pragma unroll
            for (int j = 0; j < 2; ++j)
                acc[i][j] = __builtin_amdgcn_wmma_f32_16x16x32_f16(
                    false, afrag[i], false, bfrag[j],
                    (short)0, acc[i][j], false, false);
    }

    // ---------------- epilogue ----------------------------------------------
    // C/D layout: lanes 0-15 M = r, lanes 16-31 M = r+8 ; N = lane&15
    const int row0 = bm0 + wm * 32 + ((lane >> 4) << 3);
    const int col0 = bn0 + wn * 32 + (lane & 15);

    #pragma unroll
    for (int i = 0; i < 2; ++i) {
        #pragma unroll
        for (int j = 0; j < 2; ++j) {
            const int n = col0 + j * 16;
            float bv = 0.0f;
            if constexpr (EPI == 2) { if (n < N) bv = bias[n]; }
            else                    { bv = bias[n]; }
            #pragma unroll
            for (int r = 0; r < 8; ++r) {
                const int m = row0 + i * 16 + r;
                float v = acc[i][j][r] + bv;
                if constexpr (EPI == 0) {
                    v = 1.0f / (1.0f + __expf(-v));          // sigmoid
                    v = rintf(v * 15.0f) * (1.0f / 15.0f);   // 4-bit quant (RNE)
                    ((_Float16*)Cptr)[(size_t)m * N + n] = (_Float16)v;
                } else if constexpr (EPI == 1) {
                    v = fmaxf(v, 0.0f);
                    ((_Float16*)Cptr)[(size_t)m * N + n] = (_Float16)v;
                } else {
                    if (n < N) ((float*)Cptr)[(size_t)m * N + n] = v;
                }
            }
        }
    }
}

extern "C" void kernel_launch(void* const* d_in, const int* in_sizes, int n_in,
                              void* d_out, int out_size, void* d_ws, size_t ws_size,
                              hipStream_t stream)
{
    (void)in_sizes; (void)n_in; (void)out_size; (void)ws_size;

    const float* x      = (const float*)d_in[0];
    const float* W_enc  = (const float*)d_in[1];
    const float* b_enc  = (const float*)d_in[2];
    const float* W_dec  = (const float*)d_in[3];
    const float* b_dec  = (const float*)d_in[4];
    const float* W_tail = (const float*)d_in[5];
    const float* b_tail = (const float*)d_in[6];

    const int B = 8192, D_IN = 1024, D_BOT = 256, D_HID = 1024, N_CLS = 1000;
    const int NP_TAIL = 1024;  // pad 1000 -> 1024

    // workspace layout (all f16)
    char* ws = (char*)d_ws;
    _Float16* yhat = (_Float16*)ws;                 ws += (size_t)B * D_BOT * 2;   //  4 MB
    _Float16* h    = (_Float16*)ws;                 ws += (size_t)B * D_HID * 2;   // 16 MB
    _Float16* WencF = (_Float16*)ws;                ws += (size_t)D_IN  * D_BOT  * 2;  // 512 KB
    _Float16* WdecF = (_Float16*)ws;                ws += (size_t)D_BOT * D_HID  * 2;  // 512 KB
    _Float16* WtailF = (_Float16*)ws;               ws += (size_t)D_HID * NP_TAIL * 2; //   2 MB

    const dim3 blk(256);

    // --- one-time weight repacks (f32 row-major -> f16 fragment-major) ---
    repack_w_kernel<<<dim3((D_IN  * D_BOT   + 255) / 256), blk, 0, stream>>>(
        W_enc, WencF, D_IN, D_BOT, D_BOT);
    repack_w_kernel<<<dim3((D_BOT * D_HID   + 255) / 256), blk, 0, stream>>>(
        W_dec, WdecF, D_BOT, D_HID, D_HID);
    repack_w_kernel<<<dim3((D_HID * NP_TAIL + 255) / 256), blk, 0, stream>>>(
        W_tail, WtailF, D_HID, N_CLS, NP_TAIL);

    // GEMM1: [8192,1024] x [1024,256] -> yhat (f16), sigmoid + quantize
    gemm_wmma_kernel<0, false><<<dim3(D_BOT / BN, B / BM), blk, 0, stream>>>(
        x, WencF, b_enc, yhat, D_BOT, D_BOT, D_IN);

    // GEMM2: [8192,256] x [256,1024] -> h (f16), relu
    gemm_wmma_kernel<1, true><<<dim3(D_HID / BN, B / BM), blk, 0, stream>>>(
        yhat, WdecF, b_dec, h, D_HID, D_HID, D_BOT);

    // GEMM3: [8192,1024] x [1024,1000(->1024)] -> pred (f32), store-guarded
    gemm_wmma_kernel<2, true><<<dim3(NP_TAIL / BN, B / BM), blk, 0, stream>>>(
        h, WtailF, b_tail, d_out, N_CLS, NP_TAIL, D_HID);
}
